// MLP_adap_7_layer_2_adp_73048803770368
// MI455X (gfx1250) — compile-verified, matrix operand order checked by reading, not run on hardware
//
#include <hip/hip_runtime.h>
#include <hip/hip_bf16.h>
#include <math.h>

typedef __bf16 bf16_t;
typedef __attribute__((ext_vector_type(8)))  __bf16 v8bf;
typedef __attribute__((ext_vector_type(16))) __bf16 v16bf;
typedef __attribute__((ext_vector_type(8)))  float  v8f;

#define EPSV 1e-5f

// ---------------------------------------------------------------------------
// Fragment loader: 16-bit 16x32 A/B layout (ISA 7.12.2).
// lane 0-15  : row = base + lane,    halfs = K[kb+0..7], K[kb+16..23]  (kb = kBase)
// lane 16-31 : row = base + lane-16, same but kb = kBase + 8
// ---------------------------------------------------------------------------
__device__ inline v16bf load_frag(const bf16_t* __restrict__ base, int ld,
                                  int row, int kb) {
  const bf16_t* p = base + (size_t)row * ld + kb;
  v8bf lo = *(const v8bf*)(p);
  v8bf hi = *(const v8bf*)(p + 16);
  return __builtin_shufflevector(lo, hi, 0, 1, 2, 3, 4, 5, 6, 7,
                                         8, 9, 10, 11, 12, 13, 14, 15);
}

__device__ inline v8f wmma_bf16(v16bf a, v16bf b, v8f c) {
  return __builtin_amdgcn_wmma_f32_16x16x32_bf16(false, a, false, b, (short)0,
                                                 c, false, false);
}

// ---------------------------------------------------------------------------
// Small GEMM: C[M,N] = A[M,K](bf16) * W[N,K](bf16)^T + bias  (row-major)
// Wave tile 32x32 (2x2 WMMA); block = 4 waves -> 64x64. Used for N=32/64.
// ---------------------------------------------------------------------------
__global__ __launch_bounds__(128) void gemm_bf16_nt(
    const bf16_t* __restrict__ A, const bf16_t* __restrict__ W,
    const float* __restrict__ bias, float* __restrict__ C,
    bf16_t* __restrict__ Cb, int M, int N, int K) {
  const int lane = threadIdx.x & 31;
  const int wid  = threadIdx.x >> 5;
  const int nB = blockIdx.x * 64 + (wid & 1) * 32;
  const int mB = blockIdx.y * 64 + (wid >> 1) * 32;
  if (nB >= N || mB >= M) return;

  const int r15 = lane & 15;
  const int kh  = (lane >> 4) << 3;  // 0 or 8
  const int rA0 = mB + r15, rA1 = rA0 + 16;
  const int rW0 = nB + r15, rW1 = rW0 + 16;

  v8f c00 = {}, c01 = {}, c10 = {}, c11 = {};
  for (int k = 0; k < K; k += 32) {
    if (k + 128 < K) {
      __builtin_prefetch(A + (size_t)rA0 * K + k + 128, 0, 3);
      __builtin_prefetch(W + (size_t)rW0 * K + k + 128, 0, 3);
    }
    v16bf a0 = load_frag(A, K, rA0, k + kh);
    v16bf a1 = load_frag(A, K, rA1, k + kh);
    v16bf b0 = load_frag(W, K, rW0, k + kh);
    v16bf b1 = load_frag(W, K, rW1, k + kh);
    c00 = wmma_bf16(a0, b0, c00);
    c01 = wmma_bf16(a0, b1, c01);
    c10 = wmma_bf16(a1, b0, c10);
    c11 = wmma_bf16(a1, b1, c11);
  }

  const float bs0 = bias ? bias[nB + r15] : 0.f;
  const float bs1 = bias ? bias[nB + 16 + r15] : 0.f;
  const int col0 = nB + r15;
  const int rowOff = (lane >> 4) * 8;
#pragma unroll
  for (int i = 0; i < 8; ++i) {
    const int r0 = mB + i + rowOff;
    const int r1 = mB + 16 + i + rowOff;
    const float x00 = c00[i] + bs0, x01 = c01[i] + bs1;
    const float x10 = c10[i] + bs0, x11 = c11[i] + bs1;
    C[(size_t)r0 * N + col0]      = x00;
    C[(size_t)r0 * N + col0 + 16] = x01;
    C[(size_t)r1 * N + col0]      = x10;
    C[(size_t)r1 * N + col0 + 16] = x11;
    if (Cb) {
      Cb[(size_t)r0 * N + col0]      = (bf16_t)x00;
      Cb[(size_t)r0 * N + col0 + 16] = (bf16_t)x01;
      Cb[(size_t)r1 * N + col0]      = (bf16_t)x10;
      Cb[(size_t)r1 * N + col0 + 16] = (bf16_t)x11;
    }
  }
}

// ---------------------------------------------------------------------------
// Big GEMM: wave tile 64x64 (4x4 WMMA tiles), block = 4 waves (2x2) -> 128x128.
// Requires M%128==0 && N%128==0. 16 WMMA per 8 fragment loads per k-step.
// ---------------------------------------------------------------------------
__global__ __launch_bounds__(128) void gemm_bf16_nt_big(
    const bf16_t* __restrict__ A, const bf16_t* __restrict__ W,
    const float* __restrict__ bias, float* __restrict__ C,
    bf16_t* __restrict__ Cb, int M, int N, int K) {
  const int lane = threadIdx.x & 31;
  const int wid  = threadIdx.x >> 5;
  const int nB = blockIdx.x * 128 + (wid & 1) * 64;
  const int mB = blockIdx.y * 128 + (wid >> 1) * 64;
  if (nB + 64 > N || mB + 64 > M) return;

  const int r15 = lane & 15;
  const int kh  = (lane >> 4) << 3;

  v8f acc[4][4] = {};
  for (int k = 0; k < K; k += 32) {
    // rotating prefetch: each row group touched every 4th iter, 512B ahead
    const int pg = (k >> 5) & 3;
    if (k + 256 < K) {
      __builtin_prefetch(A + (size_t)(mB + 16 * pg + r15) * K + k + 256, 0, 3);
      __builtin_prefetch(W + (size_t)(nB + 16 * pg + r15) * K + k + 256, 0, 3);
    }
    v16bf a[4], b[4];
#pragma unroll
    for (int i = 0; i < 4; ++i) {
      a[i] = load_frag(A, K, mB + 16 * i + r15, k + kh);
      b[i] = load_frag(W, K, nB + 16 * i + r15, k + kh);
    }
#pragma unroll
    for (int i = 0; i < 4; ++i)
#pragma unroll
      for (int j = 0; j < 4; ++j) acc[i][j] = wmma_bf16(a[i], b[j], acc[i][j]);
  }

  const int rowOff = (lane >> 4) * 8;
#pragma unroll
  for (int j = 0; j < 4; ++j) {
    const float bs = bias ? bias[nB + 16 * j + r15] : 0.f;
    const int col = nB + 16 * j + r15;
#pragma unroll
    for (int i = 0; i < 4; ++i) {
#pragma unroll
      for (int e = 0; e < 8; ++e) {
        const int r = mB + 16 * i + e + rowOff;
        const float x = acc[i][j][e] + bs;
        C[(size_t)r * N + col] = x;
        if (Cb) Cb[(size_t)r * N + col] = (bf16_t)x;
      }
    }
  }
}

// ---------------------------------------------------------------------------
// Column statistics over M rows: mean + variance (ddof 0 = biased, 1 = unbiased)
// ---------------------------------------------------------------------------
__global__ void colstats(const float* __restrict__ X, int M, int N,
                         float* __restrict__ mean, float* __restrict__ var,
                         int ddof) {
  const int c = blockIdx.x * 32 + threadIdx.x;
  float s = 0.f, s2 = 0.f;
  for (int r = threadIdx.y; r < M; r += 8) {
    const float v = X[(size_t)r * N + c];
    s += v;
    s2 += v * v;
  }
  __shared__ float sh_s[8][32];
  __shared__ float sh_q[8][32];
  sh_s[threadIdx.y][threadIdx.x] = s;
  sh_q[threadIdx.y][threadIdx.x] = s2;
  __syncthreads();
  if (threadIdx.y == 0) {
    float ts = 0.f, tq = 0.f;
    for (int j = 0; j < 8; ++j) {
      ts += sh_s[j][threadIdx.x];
      tq += sh_q[j][threadIdx.x];
    }
    const float m = ts / (float)M;
    mean[c] = m;
    var[c] = (tq - ts * m) / (float)(M - ddof);
  }
}

__global__ void bn_relu_apply(const float* __restrict__ Y,
                              const float* __restrict__ mean,
                              const float* __restrict__ var,
                              const float* __restrict__ g,
                              const float* __restrict__ beta,
                              float* __restrict__ Xf, bf16_t* __restrict__ Xb,
                              int M, int N) {
  const size_t i = (size_t)blockIdx.x * blockDim.x + threadIdx.x;
  if (i >= (size_t)M * N) return;
  const int c = (int)(i % N);
  float x = (Y[i] - mean[c]) * rsqrtf(var[c] + EPSV);
  x = g[c] * x + beta[c];
  x = fmaxf(x, 0.f);
  if (Xf) Xf[i] = x;
  if (Xb) Xb[i] = (bf16_t)x;
}

__global__ void ln_res_apply(const float* __restrict__ T,
                             const float* __restrict__ mean,
                             const float* __restrict__ var,
                             const float* __restrict__ gamma,
                             const float* __restrict__ bias,
                             float* __restrict__ Hf, bf16_t* __restrict__ Hb,
                             int M, int N) {
  const size_t i = (size_t)blockIdx.x * blockDim.x + threadIdx.x;
  if (i >= (size_t)M * N) return;
  const int c = (int)(i % N);
  const float x =
      gamma[c] * (T[i] - mean[c]) * rsqrtf(var[c] + EPSV) + bias[c] + Hf[i];
  Hf[i] = x;
  Hb[i] = (bf16_t)x;
}

// einsum('bi,bij->bj'): per-sample [1,64] x [64,64]; H is bf16 to halve the
// dominant HBM stream (12 reads of the [4096,4096] hyper_out).
__global__ void bvm64(const float* __restrict__ t, const bf16_t* __restrict__ H,
                      bf16_t* __restrict__ outb) {
  __shared__ float st[64];
  const int b = blockIdx.x;
  const int j = threadIdx.x;
  st[j] = t[b * 64 + j];
  __syncthreads();
  const bf16_t* Hb = H + (size_t)b * 4096;
  float acc = 0.f;
#pragma unroll
  for (int i = 0; i < 64; ++i) acc += st[i] * (float)Hb[i * 64 + j];
  outb[b * 64 + j] = (bf16_t)acc;
}

__global__ void sigmoid_b(const float* __restrict__ in,
                          bf16_t* __restrict__ outb, size_t n) {
  const size_t i = (size_t)blockIdx.x * blockDim.x + threadIdx.x;
  if (i >= n) return;
  outb[i] = (bf16_t)(1.f / (1.f + expf(-in[i])));
}

__global__ void cvt_f32_bf16(const float* __restrict__ in,
                             bf16_t* __restrict__ out, size_t n) {
  const size_t i = (size_t)blockIdx.x * blockDim.x + threadIdx.x;
  if (i < n) out[i] = (bf16_t)in[i];
}

// in[R,C] (f32, row-major) -> out[C,R] (bf16)
__global__ void transpose_cvt(const float* __restrict__ in,
                              bf16_t* __restrict__ out, int R, int C) {
  const int idx = blockIdx.x * blockDim.x + threadIdx.x;
  if (idx >= R * C) return;
  const int r = idx / C, c = idx % C;
  out[(size_t)c * R + r] = (bf16_t)in[idx];
}

__global__ void outlayer(const float* __restrict__ X, const float* __restrict__ w,
                         const float* __restrict__ b, float* __restrict__ out,
                         int N) {
  __shared__ float red[256];
  const int lane = threadIdx.x & 31;
  const int wv = threadIdx.x >> 5;
  const int row = blockIdx.x * 8 + wv;
  const float* x = X + (size_t)row * N;
  float acc = 0.f;
  for (int i = lane; i < N; i += 32) acc += x[i] * w[i];
  red[threadIdx.x] = acc;
  __syncthreads();
  for (int s = 16; s > 0; s >>= 1) {
    if (lane < s) red[threadIdx.x] += red[threadIdx.x + s];
    __syncthreads();
  }
  if (lane == 0) out[row] = 1.f / (1.f + expf(-(red[wv * 32] + b[0])));
}

__global__ void gather_kernel(const float* __restrict__ out_all,
                              const int* __restrict__ dom,
                              float* __restrict__ fin, int B) {
  const int b = blockIdx.x * blockDim.x + threadIdx.x;
  if (b < B) fin[b] = out_all[(size_t)dom[b] * B + b];
}

// ---------------------------------------------------------------------------
extern "C" void kernel_launch(void* const* d_in, const int* in_sizes, int n_in,
                              void* d_out, int out_size, void* d_ws,
                              size_t ws_size, hipStream_t stream) {
  (void)in_sizes; (void)n_in; (void)out_size; (void)ws_size;
  const int B = 4096;
  static const int DIMSA[8] = {1024, 2048, 2048, 1024, 1024, 512, 512, 256};
  static const int HIN[4] = {1024, 512, 256, 4096};

  const float* emb = (const float*)d_in[0];
  const int* dom = (const int*)d_in[1];
  const float *twW[7], *twb[7], *twg[7], *twbeta[7];
  for (int i = 0; i < 7; ++i) {
    twW[i] = (const float*)d_in[2 + 4 * i];
    twb[i] = (const float*)d_in[3 + 4 * i];
    twg[i] = (const float*)d_in[4 + 4 * i];
    twbeta[i] = (const float*)d_in[5 + 4 * i];
  }
  const float* Wout = (const float*)d_in[30];
  const float* bout = (const float*)d_in[31];
  const float *hyW[3], *hyb[3], *hyg[3], *hybeta[3];
  for (int i = 0; i < 3; ++i) {
    hyW[i] = (const float*)d_in[32 + 4 * i];
    hyb[i] = (const float*)d_in[33 + 4 * i];
    hyg[i] = (const float*)d_in[34 + 4 * i];
    hybeta[i] = (const float*)d_in[35 + 4 * i];
  }
  const float* u0 = (const float*)d_in[44];
  const float* v0 = (const float*)d_in[45];
  const float* b0 = (const float*)d_in[46];
  const float* u1 = (const float*)d_in[47];
  const float* v1 = (const float*)d_in[48];
  const float* b1 = (const float*)d_in[49];
  const float* u2 = (const float*)d_in[50];
  const float* v2 = (const float*)d_in[51];
  const float* b2 = (const float*)d_in[52];
  const float* u3 = (const float*)d_in[53];
  const float* v3 = (const float*)d_in[54];
  const float* b3 = (const float*)d_in[55];
  const float* gamma1 = (const float*)d_in[56];
  const float* bias1 = (const float*)d_in[57];
  const float* gamma2 = (const float*)d_in[58];
  const float* bias2 = (const float*)d_in[59];

  // -------- workspace carve-up --------
  char* ws = (char*)d_ws;
  size_t off = 0;
  auto alloc = [&](size_t bytes) -> void* {
    void* p = ws + off;
    off = (off + bytes + 255) & ~(size_t)255;
    return p;
  };
  bf16_t* emb_bf = (bf16_t*)alloc((size_t)B * 1024 * 2);
  bf16_t* twWb[7];
  for (int i = 0; i < 7; ++i)
    twWb[i] = (bf16_t*)alloc((size_t)3 * DIMSA[i + 1] * DIMSA[i] * 2);
  bf16_t* hyWb[3];
  for (int i = 0; i < 3; ++i)
    hyWb[i] = (bf16_t*)alloc((size_t)HIN[i + 1] * HIN[i] * 2);
  bf16_t* u0T = (bf16_t*)alloc(64 * 512 * 2);
  bf16_t* v0T = (bf16_t*)alloc(32 * 64 * 2);
  bf16_t* u1T = (bf16_t*)alloc(64 * 32 * 2);
  bf16_t* v1T = (bf16_t*)alloc(512 * 64 * 2);
  bf16_t* u2T = (bf16_t*)alloc(64 * 256 * 2);
  bf16_t* v2T = (bf16_t*)alloc(32 * 64 * 2);
  bf16_t* u3T = (bf16_t*)alloc(64 * 32 * 2);
  bf16_t* v3T = (bf16_t*)alloc(256 * 64 * 2);
  float* Y = (float*)alloc((size_t)B * 4096 * 4);
  float* Xf = (float*)alloc((size_t)B * 2048 * 4);
  bf16_t* Xb = (bf16_t*)alloc((size_t)B * 2048 * 2);
  bf16_t* Houtb = (bf16_t*)alloc((size_t)B * 4096 * 2);  // hyper_out, bf16
  float* tAf = (float*)alloc((size_t)B * 64 * 4);
  bf16_t* tBb = (bf16_t*)alloc((size_t)B * 64 * 2);
  float* Sf = (float*)alloc((size_t)B * 32 * 4);
  bf16_t* Sb = (bf16_t*)alloc((size_t)B * 32 * 2);
  float* tCf = (float*)alloc((size_t)B * 64 * 4);
  bf16_t* tDb = (bf16_t*)alloc((size_t)B * 64 * 2);
  float* meanb = (float*)alloc(4096 * 4);
  float* varb = (float*)alloc(4096 * 4);
  float* out_all = (float*)alloc((size_t)3 * B * 4);

  // -------- launch helpers --------
  auto cvt = [&](const float* in, bf16_t* out, size_t n) {
    cvt_f32_bf16<<<dim3((unsigned)((n + 255) / 256)), dim3(256), 0, stream>>>(
        in, out, n);
  };
  auto tcvt = [&](const float* in, bf16_t* out, int R, int C) {
    transpose_cvt<<<dim3((R * C + 255) / 256), dim3(256), 0, stream>>>(in, out,
                                                                       R, C);
  };
  auto gemm = [&](const bf16_t* A, const bf16_t* W, const float* bias, float* C,
                  bf16_t* Cb, int M, int N, int K) {
    if ((M % 128 == 0) && (N % 128 == 0)) {
      gemm_bf16_nt_big<<<dim3(N / 128, M / 128), dim3(128), 0, stream>>>(
          A, W, bias, C, Cb, M, N, K);
    } else {
      gemm_bf16_nt<<<dim3((N + 63) / 64, (M + 63) / 64), dim3(128), 0,
                     stream>>>(A, W, bias, C, Cb, M, N, K);
    }
  };
  auto stats = [&](const float* X, int N, int ddof) {
    colstats<<<dim3(N / 32), dim3(32, 8), 0, stream>>>(X, B, N, meanb, varb,
                                                       ddof);
  };
  auto bnrelu = [&](const float* Yp, const float* g, const float* be, float* xf,
                    bf16_t* xb, int N) {
    size_t n = (size_t)B * N;
    bn_relu_apply<<<dim3((unsigned)((n + 255) / 256)), dim3(256), 0, stream>>>(
        Yp, meanb, varb, g, be, xf, xb, B, N);
  };

  // -------- convert weights / inputs to bf16 --------
  cvt(emb, emb_bf, (size_t)B * 1024);
  for (int i = 0; i < 7; ++i)
    cvt(twW[i], twWb[i], (size_t)3 * DIMSA[i + 1] * DIMSA[i]);
  for (int i = 0; i < 3; ++i) cvt(hyW[i], hyWb[i], (size_t)HIN[i + 1] * HIN[i]);
  tcvt(u0, u0T, 512, 64);
  tcvt(v0, v0T, 64, 32);
  tcvt(u1, u1T, 32, 64);
  tcvt(v1, v1T, 64, 512);
  tcvt(u2, u2T, 256, 64);
  tcvt(v2, v2T, 64, 32);
  tcvt(u3, u3T, 32, 64);
  tcvt(v3, v3T, 64, 256);

  // -------- hyper network (shared by all domains) --------
  gemm(emb_bf, hyWb[0], hyb[0], Y, nullptr, B, 512, 1024);
  stats(Y, 512, 0);
  bnrelu(Y, hyg[0], hybeta[0], Xf, Xb, 512);
  gemm(Xb, hyWb[1], hyb[1], Y, nullptr, B, 256, 512);
  stats(Y, 256, 0);
  bnrelu(Y, hyg[1], hybeta[1], Xf, Xb, 256);
  gemm(Xb, hyWb[2], hyb[2], Y, nullptr, B, 4096, 256);
  stats(Y, 4096, 0);
  bnrelu(Y, hyg[2], hybeta[2], nullptr, Houtb, 4096);  // hyper_out [B,64,64] bf16

  auto adapter = [&](int F, const bf16_t* uaT, const bf16_t* vaT,
                     const float* ba, const bf16_t* ubT, const bf16_t* vbT,
                     const float* bb, const float* gam, const float* bia) {
    gemm(Xb, uaT, nullptr, tAf, nullptr, B, 64, F);           // h @ u
    bvm64<<<dim3(B), dim3(64), 0, stream>>>(tAf, Houtb, tBb); // x hyper_out
    gemm(tBb, vaT, ba, Sf, nullptr, B, 32, 64);               // @ v + b
    sigmoid_b<<<dim3((B * 32 + 255) / 256), dim3(256), 0, stream>>>(
        Sf, Sb, (size_t)B * 32);
    gemm(Sb, ubT, nullptr, tCf, nullptr, B, 64, 32);          // @ u2
    bvm64<<<dim3(B), dim3(64), 0, stream>>>(tCf, Houtb, tDb); // x hyper_out
    gemm(tDb, vbT, bb, Y, nullptr, B, F, 64);                 // @ v2 + b2
    stats(Y, F, 1);                                           // unbiased var
    size_t n = (size_t)B * F;
    ln_res_apply<<<dim3((unsigned)((n + 255) / 256)), dim3(256), 0, stream>>>(
        Y, meanb, varb, gam, bia, Xf, Xb, B, F);
  };

  // -------- towers --------
  for (int d = 0; d < 3; ++d) {
    const bf16_t* Ain = emb_bf;
    for (int i = 0; i < 6; ++i) {
      const int fi = DIMSA[i], fo = DIMSA[i + 1];
      gemm(Ain, twWb[i] + (size_t)d * fo * fi, twb[i] + (size_t)d * fo, Y,
           nullptr, B, fo, fi);
      stats(Y, fo, 0);
      bnrelu(Y, twg[i] + (size_t)d * fo, twbeta[i] + (size_t)d * fo, Xf, Xb, fo);
      Ain = Xb;
    }
    adapter(512, u0T, v0T, b0, u1T, v1T, b1, gamma1, bias1);
    gemm(Xb, twWb[6] + (size_t)d * 256 * 512, twb[6] + (size_t)d * 256, Y,
         nullptr, B, 256, 512);
    stats(Y, 256, 0);
    bnrelu(Y, twg[6] + (size_t)d * 256, twbeta[6] + (size_t)d * 256, Xf, Xb,
           256);
    adapter(256, u2T, v2T, b2, u3T, v3T, b3, gamma2, bias2);
    outlayer<<<dim3(B / 8), dim3(256), 0, stream>>>(
        Xf, Wout + (size_t)d * 256, bout + d, out_all + (size_t)d * B, 256);
  }

  gather_kernel<<<dim3((B + 255) / 256), dim3(256), 0, stream>>>(
      out_all, dom, (float*)d_out, B);
}
